// STNDTEncoder_83434034692803
// MI455X (gfx1250) — matrix operand order repaired and checked by reading, not compiled, not run on hardware
//
#include <hip/hip_runtime.h>
#include <hip/hip_bf16.h>

// ---- problem constants (match reference) ----
#define BB   16
#define TT   512
#define CCn  512
#define LLn  512
#define HHn  8
#define HIDN 2048
#define HSX  64
#define DSX  64

typedef __bf16 bf16_t;
typedef __attribute__((ext_vector_type(16))) __bf16 v16bf;
typedef __attribute__((ext_vector_type(8)))  __bf16 v8bf;
typedef __attribute__((ext_vector_type(8)))  float  v8f;

enum Epi { EP_F32 = 0, EP_BF16, EP_SCALE_BF16, EP_BIAS_PRELU_BF16, EP_BIAS_RES_F32, EP_RES_F32 };

// =====================================================================
// WMMA GEMM:  C[M,N] = A[M,K] * W[N,K]^T   (bf16 operands, f32 accum)
// wave tile 32x64 (2x4 of 16x16), 8 waves per block arranged WM x WN.
// Batched via blockIdx.z with (b,h) split strides (for per-head GEMMs).
// =====================================================================
template<int EP, int WM, int WN>
__global__ __launch_bounds__(256) void gemm_wmma(
    const bf16_t* __restrict__ A, int lda, long long aSb, long long aSh,
    const bf16_t* __restrict__ W, int ldw, long long wSb, long long wSh,
    void*        __restrict__ Cout, int ldc, long long cSb, long long cSh,
    const float* __restrict__ bias,
    const float* __restrict__ Res, long long rSb, long long rSh,
    int K, int Hc, float scale, const float* __restrict__ alphaPtr)
{
    const int lane = threadIdx.x & 31;
    const int wave = threadIdx.x >> 5;
    const int wm   = wave % WM;
    const int wn   = wave / WM;
    const int m0   = blockIdx.y * (WM * 32) + wm * 32;
    const int n0   = blockIdx.x * (WN * 64) + wn * 64;
    const int bz   = blockIdx.z;
    const int b    = bz / Hc;
    const int h    = bz % Hc;

    const bf16_t* Ab = A + (size_t)b * aSb + (size_t)h * aSh;
    const bf16_t* Wb = W + (size_t)b * wSb + (size_t)h * wSh;

    const int mrow = lane & 15;          // A: row within 16
    const int ncol = lane & 15;          // B: column within 16
    const int kb   = (lane >> 4) * 8;    // A: K sub-offset (ISA 16-bit A layout)
    const int kg   = (lane >> 4) * 8;    // B: K sub-offset (same layout as A)

    v8f zero = {};
    v8f acc[2][4];
#pragma unroll
    for (int mi = 0; mi < 2; ++mi)
#pragma unroll
        for (int ni = 0; ni < 4; ++ni) acc[mi][ni] = zero;

    for (int k0 = 0; k0 < K; k0 += 32) {
        v16bf af[2];
#pragma unroll
        for (int mi = 0; mi < 2; ++mi) {
            const bf16_t* ap = Ab + (size_t)(m0 + mi * 16 + mrow) * lda + k0 + kb;
            v8bf lo = *(const v8bf*)ap;          // K = kb .. kb+7
            v8bf hi = *(const v8bf*)(ap + 16);   // K = kb+16 .. kb+23
            af[mi] = __builtin_shufflevector(lo, hi,
                      0,1,2,3,4,5,6,7, 8,9,10,11,12,13,14,15);
            // unconditional speculative prefetch one cacheline ahead:
            // global_prefetch_b8; OOB speculative prefetch is dropped by HW.
            __builtin_prefetch(ap + 64, 0, 1);
        }
        v16bf wf[4];
#pragma unroll
        for (int ni = 0; ni < 4; ++ni) {
            const bf16_t* wp = Wb + (size_t)(n0 + ni * 16 + ncol) * ldw + k0 + kg;
            wf[ni] = __builtin_shufflevector(*(const v8bf*)wp, *(const v8bf*)(wp + 16),
                      0,1,2,3,4,5,6,7, 8,9,10,11,12,13,14,15);   // K = kg..kg+7, kg+16..kg+23
            __builtin_prefetch(wp + 64, 0, 1);
        }
#pragma unroll
        for (int mi = 0; mi < 2; ++mi)
#pragma unroll
            for (int ni = 0; ni < 4; ++ni)
                acc[mi][ni] = __builtin_amdgcn_wmma_f32_16x16x32_bf16(
                    false, af[mi], false, wf[ni], (short)0, acc[mi][ni], false, false);
    }

    const int   rbase = (lane >> 4) * 8;   // C/D layout: VGPR i -> M = i (+8 hi half)
    const float alpha = (EP == EP_BIAS_PRELU_BF16) ? alphaPtr[0] : 0.f;

#pragma unroll
    for (int mi = 0; mi < 2; ++mi) {
#pragma unroll
        for (int ni = 0; ni < 4; ++ni) {
#pragma unroll
            for (int i = 0; i < 8; ++i) {
                const int r = m0 + mi * 16 + rbase + i;
                const int c = n0 + ni * 16 + ncol;
                float v = acc[mi][ni][i];
                const size_t off = (size_t)b * cSb + (size_t)h * cSh + (size_t)r * ldc + c;
                if (EP == EP_F32) {
                    ((float*)Cout)[off] = v;
                } else if (EP == EP_BF16) {
                    ((bf16_t*)Cout)[off] = (bf16_t)v;
                } else if (EP == EP_SCALE_BF16) {
                    ((bf16_t*)Cout)[off] = (bf16_t)(v * scale);
                } else if (EP == EP_BIAS_PRELU_BF16) {
                    v += bias[c];
                    v = v >= 0.f ? v : alpha * v;
                    ((bf16_t*)Cout)[off] = (bf16_t)v;
                } else if (EP == EP_BIAS_RES_F32) {
                    v += bias[c] + Res[(size_t)b * rSb + (size_t)h * rSh + (size_t)r * ldc + c];
                    ((float*)Cout)[off] = v;
                } else { // EP_RES_F32
                    v += Res[(size_t)b * rSb + (size_t)h * rSh + (size_t)r * ldc + c];
                    ((float*)Cout)[off] = v;
                }
            }
        }
    }
}

// ---- LayerNorm over rows of length D (f32 in -> bf16 out) ----
__global__ __launch_bounds__(256) void ln_rows(const float* __restrict__ X,
                                               const float* __restrict__ g,
                                               const float* __restrict__ bta,
                                               bf16_t* __restrict__ Y, int D)
{
    const int row = blockIdx.x;
    const int tid = threadIdx.x;
    const float* xr = X + (size_t)row * D;
    float s = 0.f, ss = 0.f;
    for (int i = tid; i < D; i += 256) { float v = xr[i]; s += v; ss += v * v; }
    __shared__ float sh1[256], sh2[256];
    sh1[tid] = s; sh2[tid] = ss;
    __syncthreads();
    for (int st = 128; st > 0; st >>= 1) {
        if (tid < st) { sh1[tid] += sh1[tid + st]; sh2[tid] += sh2[tid + st]; }
        __syncthreads();
    }
    const float mean = sh1[0] / D;
    const float var  = sh2[0] / D - mean * mean;
    const float inv  = rsqrtf(var + 1e-5f);
    for (int i = tid; i < D; i += 256) {
        float v = (xr[i] - mean) * inv * g[i] + bta[i];
        Y[(size_t)row * D + i] = (bf16_t)v;
    }
}

// ---- softmax over rows of 512 bf16, in place ----
__global__ __launch_bounds__(256) void softmax_rows(bf16_t* __restrict__ P)
{
    const int row = blockIdx.x;
    const int tid = threadIdx.x;
    bf16_t* pr = P + (size_t)row * 512;
    float v0 = (float)pr[tid];
    float v1 = (float)pr[tid + 256];
    __shared__ float sh[256];
    sh[tid] = fmaxf(v0, v1);
    __syncthreads();
    for (int st = 128; st > 0; st >>= 1) {
        if (tid < st) sh[tid] = fmaxf(sh[tid], sh[tid + st]);
        __syncthreads();
    }
    const float m = sh[0];
    __syncthreads();
    float e0 = __expf(v0 - m), e1 = __expf(v1 - m);
    sh[tid] = e0 + e1;
    __syncthreads();
    for (int st = 128; st > 0; st >>= 1) {
        if (tid < st) sh[tid] += sh[tid + st];
        __syncthreads();
    }
    const float inv = 1.f / sh[0];
    pr[tid]       = (bf16_t)(e0 * inv);
    pr[tid + 256] = (bf16_t)(e1 * inv);
}

// ---- V [B,T,H,HS] -> Vt [B,H,HS,T] (bf16) ----
__global__ __launch_bounds__(256) void transpose_v(const bf16_t* __restrict__ V,
                                                   bf16_t* __restrict__ Vt)
{
    size_t gid = (size_t)blockIdx.x * 256 + threadIdx.x;   // over B*H*HS*T
    const int t = gid & (TT - 1);
    const int k = (gid >> 9) & (HSX - 1);
    const int h = (gid >> 15) & (HHn - 1);
    const int b = (int)(gid >> 18);
    Vt[gid] = V[(((size_t)b * TT + t) * HHn + h) * HSX + k];
}

// ---- mean over heads: P [B,H,C,C] bf16 -> SW f32 (d_out) + SW bf16 ----
__global__ __launch_bounds__(256) void mean_heads(const bf16_t* __restrict__ P,
                                                  float* __restrict__ SWf,
                                                  bf16_t* __restrict__ SWb)
{
    size_t gid = (size_t)blockIdx.x * 256 + threadIdx.x;   // over B*C*C
    const size_t cc = (size_t)CCn * CCn;
    const size_t b  = gid / cc;
    const size_t r  = gid - b * cc;
    float s = 0.f;
#pragma unroll
    for (int h = 0; h < HHn; ++h) s += (float)P[(b * HHn + h) * cc + r];
    s *= (1.f / HHn);
    SWf[gid] = s;
    SWb[gid] = (bf16_t)s;
}

// ---- f32 -> bf16 conversion ----
__global__ __launch_bounds__(256) void cvt_bf16(const float* __restrict__ X,
                                                bf16_t* __restrict__ Y, int n)
{
    int gid = blockIdx.x * 256 + threadIdx.x;
    if (gid < n) Y[gid] = (bf16_t)X[gid];
}

// =====================================================================
#define GEMM(EPv, WMv, WNv, Aptr, ldaV, aSbV, aShV, Wptr, ldwV, wSbV, wShV,      \
             Cptr, ldcV, cSbV, cShV, biasP, ResP, rSbV, rShV, Mv, Nv, Kv, NBv,   \
             HcV, scV)                                                           \
    gemm_wmma<EPv, WMv, WNv><<<dim3((Nv) / ((WNv) * 64), (Mv) / ((WMv) * 32),    \
                                    (NBv)), 256, 0, stream>>>(                   \
        Aptr, ldaV, aSbV, aShV, Wptr, ldwV, wSbV, wShV, Cptr, ldcV, cSbV, cShV,  \
        biasP, ResP, rSbV, rShV, Kv, HcV, scV, alphaP)

extern "C" void kernel_launch(void* const* d_in, const int* in_sizes, int n_in,
                              void* d_out, int out_size, void* d_ws, size_t ws_size,
                              hipStream_t stream)
{
    (void)in_sizes; (void)n_in; (void)out_size; (void)ws_size;
    const float* src    = (const float*)d_in[0];
    const float* ssrc   = (const float*)d_in[1];
    const float* Wq_t   = (const float*)d_in[2];
    const float* Wk_t   = (const float*)d_in[3];
    const float* Wv_t   = (const float*)d_in[4];
    const float* Wo_t   = (const float*)d_in[5];
    const float* Wq_s   = (const float*)d_in[6];
    const float* Wk_s   = (const float*)d_in[7];
    const float* W1     = (const float*)d_in[8];
    const float* b1     = (const float*)d_in[9];
    const float* W2     = (const float*)d_in[10];
    const float* b2     = (const float*)d_in[11];
    const float* tsW1   = (const float*)d_in[12];
    const float* tsb1   = (const float*)d_in[13];
    const float* tsW2   = (const float*)d_in[14];
    const float* tsb2   = (const float*)d_in[15];
    const float* alphaP = (const float*)d_in[16];
    const float* g1     = (const float*)d_in[17];
    const float* be1    = (const float*)d_in[18];
    const float* g2     = (const float*)d_in[19];
    const float* be2    = (const float*)d_in[20];
    const float* tsg1   = (const float*)d_in[21];
    const float* tsbe1  = (const float*)d_in[22];
    const float* tsg2   = (const float*)d_in[23];
    const float* tsbe2  = (const float*)d_in[24];
    const float* sg     = (const float*)d_in[25];
    const float* sbe    = (const float*)d_in[26];

    const int BT = BB * TT;                 // 8192 rows (also B*C for spatial)
    float* out_ts = (float*)d_out;          // [B,T,C]
    float* out_sw = out_ts + (size_t)BB * TT * CCn;  // [B,C,C]

    // ---- workspace carve-up ----
    char* wsp = (char*)d_ws;
    auto alloc = [&](size_t bytes) -> char* {
        char* p = wsp;
        wsp += (bytes + 255) & ~(size_t)255;
        return p;
    };
    bf16_t* wq    = (bf16_t*)alloc((size_t)CCn * CCn * 2);
    bf16_t* wk    = (bf16_t*)alloc((size_t)CCn * CCn * 2);
    bf16_t* wv    = (bf16_t*)alloc((size_t)CCn * CCn * 2);
    bf16_t* wo    = (bf16_t*)alloc((size_t)CCn * CCn * 2);
    bf16_t* wqs   = (bf16_t*)alloc((size_t)LLn * LLn * 2);
    bf16_t* wks   = (bf16_t*)alloc((size_t)LLn * LLn * 2);
    bf16_t* w1b   = (bf16_t*)alloc((size_t)HIDN * CCn * 2);
    bf16_t* w2b   = (bf16_t*)alloc((size_t)CCn * HIDN * 2);
    bf16_t* tw1b  = (bf16_t*)alloc((size_t)HIDN * CCn * 2);
    bf16_t* tw2b  = (bf16_t*)alloc((size_t)CCn * HIDN * 2);
    bf16_t* xbf   = (bf16_t*)alloc((size_t)BT * CCn * 2);     // LN outputs (reused)
    bf16_t* Qb    = (bf16_t*)alloc((size_t)BT * CCn * 2);     // also QS
    bf16_t* Kb    = (bf16_t*)alloc((size_t)BT * CCn * 2);     // also KS
    bf16_t* Vb    = (bf16_t*)alloc((size_t)BT * CCn * 2);
    bf16_t* Vt    = (bf16_t*)alloc((size_t)BT * CCn * 2);
    bf16_t* attnb = (bf16_t*)alloc((size_t)BT * CCn * 2);
    bf16_t* hbuf  = (bf16_t*)alloc((size_t)BT * HIDN * 2);    // FFN hidden (reused)
    bf16_t* Pb    = (bf16_t*)alloc((size_t)BB * HHn * TT * TT * 2);  // probs (reused)
    bf16_t* swbf  = (bf16_t*)alloc((size_t)BB * CCn * CCn * 2);
    float*  src1  = (float*)alloc((size_t)BT * CCn * 4);      // residual streams
    float*  src2  = (float*)alloc((size_t)BT * CCn * 4);

    // ---- weights -> bf16 ----
    auto CVT = [&](const float* x, bf16_t* y, int n) {
        cvt_bf16<<<(n + 255) / 256, 256, 0, stream>>>(x, y, n);
    };
    CVT(Wq_t, wq, CCn * CCn);   CVT(Wk_t, wk, CCn * CCn);
    CVT(Wv_t, wv, CCn * CCn);   CVT(Wo_t, wo, CCn * CCn);
    CVT(Wq_s, wqs, LLn * LLn);  CVT(Wk_s, wks, LLn * LLn);
    CVT(W1, w1b, HIDN * CCn);   CVT(W2, w2b, CCn * HIDN);
    CVT(tsW1, tw1b, HIDN * CCn); CVT(tsW2, tw2b, CCn * HIDN);

    const long long ZL = 0;

    // ---- temporal attention ----
    ln_rows<<<BT, 256, 0, stream>>>(src, g1, be1, xbf, CCn);
    GEMM(EP_BF16, 2, 4, xbf, CCn, ZL, ZL, wq, CCn, ZL, ZL, Qb, CCn, ZL, ZL,
         nullptr, nullptr, ZL, ZL, BT, CCn, CCn, 1, 1, 1.f);
    GEMM(EP_BF16, 2, 4, xbf, CCn, ZL, ZL, wk, CCn, ZL, ZL, Kb, CCn, ZL, ZL,
         nullptr, nullptr, ZL, ZL, BT, CCn, CCn, 1, 1, 1.f);
    GEMM(EP_BF16, 2, 4, xbf, CCn, ZL, ZL, wv, CCn, ZL, ZL, Vb, CCn, ZL, ZL,
         nullptr, nullptr, ZL, ZL, BT, CCn, CCn, 1, 1, 1.f);
    transpose_v<<<(BB * TT * CCn) / 256, 256, 0, stream>>>(Vb, Vt);

    // scores[b,h] = (Q_h K_h^T) / 8, bf16
    GEMM(EP_SCALE_BF16, 2, 4,
         Qb, CCn, (long long)TT * CCn, (long long)HSX,
         Kb, CCn, (long long)TT * CCn, (long long)HSX,
         Pb, TT, (long long)HHn * TT * TT, (long long)TT * TT,
         nullptr, nullptr, ZL, ZL, TT, TT, HSX, BB * HHn, HHn, 0.125f);
    softmax_rows<<<BB * HHn * TT, 256, 0, stream>>>(Pb);

    // attn[b,h] = P Vt^T  (N = 64, so waves 8x1)
    GEMM(EP_BF16, 8, 1,
         Pb, TT, (long long)HHn * TT * TT, (long long)TT * TT,
         Vt, TT, (long long)HHn * HSX * TT, (long long)HSX * TT,
         attnb, CCn, (long long)TT * CCn, (long long)HSX,
         nullptr, nullptr, ZL, ZL, TT, HSX, TT, BB * HHn, HHn, 1.f);

    // out-proj + residual
    GEMM(EP_RES_F32, 2, 4, attnb, CCn, ZL, ZL, wo, CCn, ZL, ZL, src1, CCn, ZL, ZL,
         nullptr, src, ZL, ZL, BT, CCn, CCn, 1, 1, 1.f);

    // ---- FFN 1 ----
    ln_rows<<<BT, 256, 0, stream>>>(src1, g2, be2, xbf, CCn);
    GEMM(EP_BIAS_PRELU_BF16, 2, 4, xbf, CCn, ZL, ZL, w1b, CCn, ZL, ZL,
         hbuf, HIDN, ZL, ZL, b1, nullptr, ZL, ZL, BT, HIDN, CCn, 1, 1, 1.f);
    GEMM(EP_BIAS_RES_F32, 2, 4, hbuf, HIDN, ZL, ZL, w2b, HIDN, ZL, ZL,
         src2, CCn, ZL, ZL, b2, src1, ZL, ZL, BT, CCn, HIDN, 1, 1, 1.f);

    // ---- spatial attention weights ----
    ln_rows<<<BB * CCn, 256, 0, stream>>>(ssrc, sg, sbe, xbf, LLn);
    GEMM(EP_BF16, 2, 4, xbf, LLn, ZL, ZL, wqs, LLn, ZL, ZL, Qb, LLn, ZL, ZL,
         nullptr, nullptr, ZL, ZL, BT, LLn, LLn, 1, 1, 1.f);
    GEMM(EP_BF16, 2, 4, xbf, LLn, ZL, ZL, wks, LLn, ZL, ZL, Kb, LLn, ZL, ZL,
         nullptr, nullptr, ZL, ZL, BT, LLn, LLn, 1, 1, 1.f);
    GEMM(EP_SCALE_BF16, 2, 4,
         Qb, LLn, (long long)CCn * LLn, (long long)DSX,
         Kb, LLn, (long long)CCn * LLn, (long long)DSX,
         Pb, CCn, (long long)HHn * CCn * CCn, (long long)CCn * CCn,
         nullptr, nullptr, ZL, ZL, CCn, CCn, DSX, BB * HHn, HHn, 0.125f);
    softmax_rows<<<BB * HHn * CCn, 256, 0, stream>>>(Pb);
    mean_heads<<<(BB * CCn * CCn) / 256, 256, 0, stream>>>(Pb, out_sw, swbf);

    // ---- temporal-spatial mixing ----
    ln_rows<<<BT, 256, 0, stream>>>(src2, tsg1, tsbe1, xbf, CCn);
    GEMM(EP_RES_F32, 2, 4,
         xbf, CCn, (long long)TT * CCn, ZL,
         swbf, CCn, (long long)CCn * CCn, ZL,
         src1, CCn, (long long)TT * CCn, ZL,
         nullptr, src2, (long long)TT * CCn, ZL, TT, CCn, CCn, BB, 1, 1.f);

    // ---- FFN 2 (writes ts_out straight to d_out) ----
    ln_rows<<<BT, 256, 0, stream>>>(src1, tsg2, tsbe2, xbf, CCn);
    GEMM(EP_BIAS_PRELU_BF16, 2, 4, xbf, CCn, ZL, ZL, tw1b, CCn, ZL, ZL,
         hbuf, HIDN, ZL, ZL, tsb1, nullptr, ZL, ZL, BT, HIDN, CCn, 1, 1, 1.f);
    GEMM(EP_BIAS_RES_F32, 2, 4, hbuf, HIDN, ZL, ZL, tw2b, HIDN, ZL, ZL,
         out_ts, CCn, ZL, ZL, tsb2, src1, ZL, ZL, BT, CCn, HIDN, 1, 1, 1.f);
}
